// GCNModel_51642686767945
// MI455X (gfx1250) — matrix operand order repaired and checked by reading, not compiled, not run on hardware
//
#include <hip/hip_runtime.h>
#include <hip/hip_bf16.h>

typedef __attribute__((ext_vector_type(16))) _Float16 v16h;
typedef __attribute__((ext_vector_type(8)))  _Float16 v8h;
typedef __attribute__((ext_vector_type(8)))  float    v8f;

#define IN_DIM 128
#define HID    256
#define N_CLS  40
#define NPAD3  64     // layer-3 padded output width (40 -> 64)

// ---------------------------------------------------------------------------
// Degree / normalization kernels
// ---------------------------------------------------------------------------
__global__ __launch_bounds__(256) void init_deg_kernel(float* deg, int n) {
    int i = blockIdx.x * blockDim.x + threadIdx.x;
    if (i < n) deg[i] = 1.0f;               // self-loop contributes 1
}

__global__ __launch_bounds__(256) void deg_count_kernel(const long long* __restrict__ dst,
                                                        float* deg, int e) {
    int i = blockIdx.x * blockDim.x + threadIdx.x;
    if (i < e) atomicAdd(&deg[(int)dst[i]], 1.0f);
}

__global__ __launch_bounds__(256) void dinv_kernel(float* deg, int n) {
    int i = blockIdx.x * blockDim.x + threadIdx.x;
    if (i < n) deg[i] = rsqrtf(deg[i]);     // deg >= 1 always (self loops)
}

// ---------------------------------------------------------------------------
// Weight pre-pack: Wt[n][k] = f16(W[k][n]), zero-padded for n >= N.
// Transposed layout makes each WMMA B-fragment a contiguous 32-byte run.
// ---------------------------------------------------------------------------
__global__ __launch_bounds__(256) void pack_w_kernel(const float* __restrict__ W,
                                                     _Float16* __restrict__ Wt,
                                                     int K, int N, int Npad) {
    int idx = blockIdx.x * blockDim.x + threadIdx.x;
    if (idx >= Npad * K) return;
    int n = idx / K;
    int k = idx - n * K;
    Wt[idx] = (n < N) ? (_Float16)W[(long)k * N + n] : (_Float16)0.0f;
}

// ---------------------------------------------------------------------------
// WMMA GEMM: H[M,N] = A[M,K] @ W[K,N], W pre-packed f16-transposed [Npad][K].
// One wave -> 16x64 output strip (A-frag reused over 4 B-frags).
// Fully unrolled K with 1-step software pipeline: next step's loads are
// issued before the current step's WMMAs so s_wait overlaps with math.
// RELU (template) folds the previous layer's ReLU into the A fragment as
// packed f16 max (v_pk_max_num_f16), after v_cvt_pk_f16_f32 conversion.
// ---------------------------------------------------------------------------
template <int KK, bool RELU>
__global__ __launch_bounds__(256) void gemm_wmma_kernel(const float* __restrict__ A,
                                                        const _Float16* __restrict__ Wt,
                                                        float* __restrict__ H,
                                                        int M, int N) {
    const int lane = threadIdx.x & 31;
    const int wave = threadIdx.x >> 5;
    const int m0   = blockIdx.x * 128 + wave * 16;
    const int n0   = blockIdx.y * 64;
    if (m0 >= M) return;                     // wave-uniform: EXEC stays all-ones

    const int half16 = lane >> 4;            // 0 for lanes 0-15, 1 for 16-31
    const int l16    = lane & 15;

    v8f acc[4];
    #pragma unroll
    for (int t = 0; t < 4; ++t) acc[t] = (v8f){0,0,0,0,0,0,0,0};

    // A row = lane%16; elems 0-15 <- K=kb+16*half16 .. +15 (contiguous)
    const float* __restrict__ Arow = A + (long)(m0 + l16) * KK + half16 * 16;
    // B: lane holds column n0+16t+l16; elems = 16 consecutive K (32B aligned).
    const _Float16* __restrict__ Bcol[4];
    #pragma unroll
    for (int t = 0; t < 4; ++t)
        Bcol[t] = Wt + (long)(n0 + t * 16 + l16) * KK + half16 * 16;

    constexpr int NSTEP = KK / 32;

    // ---- prologue: load step 0 ----
    v8f  alo = *(const v8f*)(Arow);          // K = kb .. kb+7
    v8f  ahi = *(const v8f*)(Arow + 8);      // K = kb+8 .. kb+15
    v16h b[4];
    #pragma unroll
    for (int t = 0; t < 4; ++t) b[t] = *(const v16h*)(Bcol[t]);

    #pragma unroll
    for (int s = 0; s < NSTEP; ++s) {
        // ---- issue next step's loads before this step's math ----
        v8f  alo_n, ahi_n;
        v16h b_n[4];
        if (s + 1 < NSTEP) {
            const int k0 = (s + 1) * 32;
            alo_n = *(const v8f*)(Arow + k0);
            ahi_n = *(const v8f*)(Arow + k0 + 8);
            #pragma unroll
            for (int t = 0; t < 4; ++t) b_n[t] = *(const v16h*)(Bcol[t] + k0);
        }
        // ---- build A fragment: pk converts + optional packed-f16 ReLU ----
        v8h lo8 = __builtin_convertvector(alo, v8h);
        v8h hi8 = __builtin_convertvector(ahi, v8h);
        v16h a = __builtin_shufflevector(lo8, hi8,
                     0, 1, 2, 3, 4, 5, 6, 7, 8, 9, 10, 11, 12, 13, 14, 15);
        if (RELU) {
            const v16h z16 = {};
            a = __builtin_elementwise_max(a, z16);   // v_pk_max_num_f16
        }
        // ---- 4 WMMAs on the current fragments ----
        #pragma unroll
        for (int t = 0; t < 4; ++t)
            acc[t] = __builtin_amdgcn_wmma_f32_16x16x32_f16(
                         false, a, false, b[t], (short)0, acc[t], false, false);
        // ---- rotate pipeline registers ----
        if (s + 1 < NSTEP) {
            alo = alo_n; ahi = ahi_n;
            #pragma unroll
            for (int t = 0; t < 4; ++t) b[t] = b_n[t];
        }
    }

    // ---- D store: VGPR g -> row m0 + g + 8*half16, col = n0+16t+lane%16 ----
    const int rbase = m0 + half16 * 8;
    #pragma unroll
    for (int t = 0; t < 4; ++t) {
        const int col = n0 + t * 16 + l16;
        if (col < N) {                        // epilogue only; WMMAs already done
            #pragma unroll
            for (int g = 0; g < 8; ++g)
                H[(long)(rbase + g) * N + col] = acc[t][g];
        }
    }
}

// ---------------------------------------------------------------------------
// out[i,:] = bias + dinv[i]^2 * h[i,:]   (self-loop term; also initializes out)
// ---------------------------------------------------------------------------
__global__ __launch_bounds__(256) void self_bias_kernel(const float* __restrict__ h,
                                                        const float* __restrict__ dinv,
                                                        const float* __restrict__ bias,
                                                        float* __restrict__ out,
                                                        int n, int d) {
    long idx = (long)blockIdx.x * blockDim.x + threadIdx.x;
    long tot = (long)n * d;
    if (idx < tot) {
        int i = (int)(idx / d);
        int j = (int)(idx - (long)i * d);
        float di = dinv[i];
        out[idx] = bias[j] + di * di * h[idx];
    }
}

// ---------------------------------------------------------------------------
// Edge scatter: one wave per edge, lanes stride the feature dim.
// out[dst,:] += dinv[src]*dinv[dst] * h[src,:]  via global_atomic_add_f32
// ---------------------------------------------------------------------------
__global__ __launch_bounds__(256) void edge_agg_kernel(const long long* __restrict__ src,
                                                       const long long* __restrict__ dst,
                                                       const float* __restrict__ dinv,
                                                       const float* __restrict__ h,
                                                       float* __restrict__ out,
                                                       int e, int d) {
    int eid = blockIdx.x * (blockDim.x >> 5) + (threadIdx.x >> 5);
    if (eid >= e) return;
    const int lane = threadIdx.x & 31;
    const int s = (int)src[eid];
    const int t = (int)dst[eid];
    const float w = dinv[s] * dinv[t];
    const float* __restrict__ hs = h + (long)s * d;
    float* __restrict__ od = out + (long)t * d;
    for (int j = lane; j < d; j += 32)
        atomicAdd(&od[j], w * hs[j]);
}

// ---------------------------------------------------------------------------
// Row-wise log-softmax over N_CLS=40 classes, one thread per node.
// ---------------------------------------------------------------------------
__global__ __launch_bounds__(256) void log_softmax_kernel(const float* __restrict__ in,
                                                          float* __restrict__ out, int n) {
    int i = blockIdx.x * blockDim.x + threadIdx.x;
    if (i >= n) return;
    const float* __restrict__ row = in + (long)i * N_CLS;
    float v[N_CLS];
    float m = -3.4e38f;
    #pragma unroll
    for (int j = 0; j < N_CLS; ++j) { v[j] = row[j]; m = fmaxf(m, v[j]); }
    float s = 0.0f;
    #pragma unroll
    for (int j = 0; j < N_CLS; ++j) s += __expf(v[j] - m);
    const float lse = m + __logf(s);
    float* __restrict__ orow = out + (long)i * N_CLS;
    #pragma unroll
    for (int j = 0; j < N_CLS; ++j) orow[j] = v[j] - lse;
}

// ---------------------------------------------------------------------------
// Host-side launch sequence (graph-capture safe: only kernel launches).
// ---------------------------------------------------------------------------
extern "C" void kernel_launch(void* const* d_in, const int* in_sizes, int n_in,
                              void* d_out, int out_size, void* d_ws, size_t ws_size,
                              hipStream_t stream) {
    const float*     x  = (const float*)d_in[0];
    const long long* ei = (const long long*)d_in[1];   // int64 per reference, [2, E]
    const float*     W1 = (const float*)d_in[2];
    const float*     b1 = (const float*)d_in[3];
    const float*     W2 = (const float*)d_in[4];
    const float*     b2 = (const float*)d_in[5];
    const float*     W3 = (const float*)d_in[6];
    const float*     b3 = (const float*)d_in[7];
    float* out = (float*)d_out;

    const int N = in_sizes[0] / IN_DIM;   // 100000
    const int E = in_sizes[1] / 2;        // 1600000
    const long long* esrc = ei;
    const long long* edst = ei + E;

    // Workspace: dinv[N] | hbuf[N*HID] | abuf[N*HID] | packed f16 weights
    float* dinv = (float*)d_ws;
    float* hbuf = dinv + N;
    float* abuf = hbuf + (size_t)N * HID;
    _Float16* w1t = (_Float16*)(abuf + (size_t)N * HID);  // [HID  ][IN_DIM]
    _Float16* w2t = w1t + (size_t)HID * IN_DIM;           // [HID  ][HID]
    _Float16* w3t = w2t + (size_t)HID * HID;              // [NPAD3][HID]

    const int TB = 256;
    const dim3 blk(TB);

    // ---- weight pre-pack (tiny) ----
    pack_w_kernel<<<(HID * IN_DIM + TB - 1) / TB, blk, 0, stream>>>(W1, w1t, IN_DIM, HID, HID);
    pack_w_kernel<<<(HID * HID + TB - 1) / TB, blk, 0, stream>>>(W2, w2t, HID, HID, HID);
    pack_w_kernel<<<(NPAD3 * HID + TB - 1) / TB, blk, 0, stream>>>(W3, w3t, HID, N_CLS, NPAD3);

    // ---- normalization: deg -> dinv (in place) ----
    init_deg_kernel<<<(N + TB - 1) / TB, blk, 0, stream>>>(dinv, N);
    deg_count_kernel<<<(E + TB - 1) / TB, blk, 0, stream>>>(edst, dinv, E);
    dinv_kernel<<<(N + TB - 1) / TB, blk, 0, stream>>>(dinv, N);

    const int mblocks = (N + 127) / 128;
    const int eblocks = (E + 7) / 8;                  // 8 waves (edges) per block

    // ---- Layer 1: x[N,128] @ W1[128,256] ----
    gemm_wmma_kernel<IN_DIM, false><<<dim3(mblocks, HID / 64), blk, 0, stream>>>(x, w1t, hbuf, N, HID);
    self_bias_kernel<<<(int)(((long)N * HID + TB - 1) / TB), blk, 0, stream>>>(hbuf, dinv, b1, abuf, N, HID);
    edge_agg_kernel<<<eblocks, blk, 0, stream>>>(esrc, edst, dinv, hbuf, abuf, E, HID);

    // ---- Layer 2: relu(abuf)[N,256] @ W2[256,256]  (ReLU folded into GEMM) ----
    gemm_wmma_kernel<HID, true><<<dim3(mblocks, HID / 64), blk, 0, stream>>>(abuf, w2t, hbuf, N, HID);
    self_bias_kernel<<<(int)(((long)N * HID + TB - 1) / TB), blk, 0, stream>>>(hbuf, dinv, b2, abuf, N, HID);
    edge_agg_kernel<<<eblocks, blk, 0, stream>>>(esrc, edst, dinv, hbuf, abuf, E, HID);

    // ---- Layer 3: relu(abuf)[N,256] @ W3[256,40] (padded to 64) ----
    gemm_wmma_kernel<HID, true><<<dim3(mblocks, 1), blk, 0, stream>>>(abuf, w3t, hbuf, N, N_CLS);
    float* a3 = abuf;  // abuf consumed by the GEMM read; safe to overwrite now
    self_bias_kernel<<<(int)(((long)N * N_CLS + TB - 1) / TB), blk, 0, stream>>>(hbuf, dinv, b3, a3, N, N_CLS);
    edge_agg_kernel<<<eblocks, blk, 0, stream>>>(esrc, edst, dinv, hbuf, a3, E, N_CLS);

    // ---- log-softmax -> d_out ----
    log_softmax_kernel<<<(N + TB - 1) / TB, blk, 0, stream>>>(a3, out, N);
}
